// Shape2DPainterNet_34694745817734
// MI455X (gfx1250) — matrix operand order repaired and checked by reading, not compiled, not run on hardware
//
#include <hip/hip_runtime.h>
#include <hip/hip_bf16.h>
#include <stdint.h>

// Shape2DPainterNet on MI455X (gfx1250):
//   Kernel 1: fused embedding-gather + 64-step LSTM using v_wmma_f32_16x16x32_bf16.
//             - each wave owns a 16-row batch tile for all 64 timesteps
//             - c-state in VGPRs, h republished to LDS as bf16 A-fragments each step
//             - [W_ih;W_hh] (128x256 bf16, K-major) resident in LDS for B-fragments
//             - next timestep's embedding rows DMA'd global->LDS with
//               GLOBAL_LOAD_ASYNC_TO_LDS_B64 (ASYNCcnt), double-buffered against WMMA
//   Kernel 2: per-row heads + Gumbel-max sampling + rel/abs location join.

#define T_LEN 64
#define D_DIM 64
#define G_DIM 256   // 4*D
#define PADK 134    // 67 DWORDs row stride: odd, coprime with 64 banks -> conflict-free

typedef __attribute__((ext_vector_type(16))) __bf16 v16bf;
typedef __attribute__((ext_vector_type(8)))  float  v8f;

__device__ __forceinline__ float sigf(float x) {
    return 1.0f / (1.0f + __expf(-x));
}
__device__ __forceinline__ float tanhfast(float x) {
    float e = __expf(2.0f * x);
    return (e - 1.0f) / (e + 1.0f);
}

// LDS byte offset of a shared-memory pointer (flat LDS aperture: low 32 bits = offset)
__device__ __forceinline__ uint32_t lds_off(const void* p) {
    return (uint32_t)(uintptr_t)p;
}

// ---------------------------------------------------------------------------
// Kernel 1: LSTM over T=64 with WMMA. block = 128 threads = 4 waves,
// each wave handles 16 batch rows; grid = B/64 blocks.
// ---------------------------------------------------------------------------
__global__ __launch_bounds__(128) void lstm_wmma_kernel(
    const int*   __restrict__ inst,    // [B, T]
    const float* __restrict__ embed,   // [V+1, 64]
    const float* __restrict__ W_ih,    // [64, 256]
    const float* __restrict__ W_hh,    // [64, 256]
    const float* __restrict__ b_ih,    // [256]
    const float* __restrict__ b_hh,    // [256]
    float*       __restrict__ e_out)   // [B, 64]
{
    // Transposed combined weights: WldsT[n][k], k in [0,128): 0-63 = W_ih, 64-127 = W_hh.
    __shared__ __bf16 WldsT[G_DIM][PADK];     // 256*134*2 = 68608 B
    __shared__ __bf16 XH[4][16][PADK];        // per-wave [row][k]: k<64 = x_t, 64..127 = h
    __shared__ float  stage[4][16][64];       // per-wave async-DMA staging (f32 embeds)
    __shared__ int    tokl[4][16 * 64];       // per-wave token tile
    __shared__ float  bsum[G_DIM];            // b_ih + b_hh

    const int tid     = threadIdx.x;
    const int lane    = tid & 31;
    const int w       = tid >> 5;
    const int rowbase = blockIdx.x * 64 + w * 16;
    const int colq    = lane & 15;
    const int khalf   = lane >> 4;

    // --- load weights (transposed, bf16) ---
    for (int idx = tid; idx < 128 * G_DIM; idx += 128) {
        int k = idx >> 8;           // 0..127
        int n = idx & 255;          // 0..255 (coalesced)
        float v = (k < 64) ? W_ih[k * G_DIM + n] : W_hh[(k - 64) * G_DIM + n];
        WldsT[n][k] = (__bf16)v;
    }
    for (int idx = tid; idx < G_DIM; idx += 128)
        bsum[idx] = b_ih[idx] + b_hh[idx];

    // --- preload this wave's tokens ---
    for (int idx = lane; idx < 16 * 64; idx += 32) {
        int r = idx >> 6, t = idx & 63;
        tokl[w][idx] = inst[(size_t)(rowbase + r) * T_LEN + t];
    }
    // --- zero h region of XH ---
    for (int idx = lane; idx < 16 * 64; idx += 32)
        XH[w][idx >> 6][64 + (idx & 63)] = (__bf16)0.0f;

    __syncthreads();

    // --- prologue: async-DMA t=0 embedding rows into staging (ASYNCcnt path) ---
    #pragma unroll 4
    for (int r = 0; r < 16; ++r) {
        int tok = tokl[w][(r << 6) | 0];
        uint64_t ga = (uint64_t)(uintptr_t)(embed + (size_t)tok * D_DIM + 2 * lane);
        uint32_t lo = lds_off(&stage[w][r][2 * lane]);
        asm volatile("global_load_async_to_lds_b64 %0, %1, off"
                     :: "v"(lo), "v"(ga) : "memory");
    }

    float cst[4][8];    // c-state: 4 d-groups x 8 rows-per-lane
    float hnew[4][8];
    #pragma unroll
    for (int j = 0; j < 4; ++j)
        #pragma unroll
        for (int v = 0; v < 8; ++v) { cst[j][v] = 0.0f; hnew[j][v] = 0.0f; }

    for (int t = 0; t < T_LEN; ++t) {
        // ---- wait for this step's DMA, convert staging f32 -> bf16 into XH ----
        asm volatile("s_wait_asynccnt 0x0" ::: "memory");
        #pragma unroll 4
        for (int r = 0; r < 16; ++r) {
            float2 xv = *reinterpret_cast<const float2*>(&stage[w][r][2 * lane]);
            XH[w][r][2 * lane]     = (__bf16)xv.x;
            XH[w][r][2 * lane + 1] = (__bf16)xv.y;
        }
        // staging reads must have landed before the engine may overwrite them
        asm volatile("s_wait_dscnt 0x0" ::: "memory");

        // ---- kick off next step's embedding DMA; overlaps with WMMA below ----
        if (t + 1 < T_LEN) {
            #pragma unroll 4
            for (int r = 0; r < 16; ++r) {
                int tok = tokl[w][(r << 6) | (t + 1)];
                uint64_t ga = (uint64_t)(uintptr_t)(embed + (size_t)tok * D_DIM + 2 * lane);
                uint32_t lo = lds_off(&stage[w][r][2 * lane]);
                asm volatile("global_load_async_to_lds_b64 %0, %1, off"
                             :: "v"(lo), "v"(ga) : "memory");
            }
        }

        // ---- A fragments: [x h] rows, K=128 in 4 chunks of 32 (j-invariant) ----
        v16bf afr[4];
        #pragma unroll
        for (int kk = 0; kk < 4; ++kk) {
            #pragma unroll
            for (int v = 0; v < 8; ++v) {
                int kb = kk * 32 + khalf * 8 + ((v < 4) ? 2 * v : 16 + 2 * (v - 4));
                afr[kk][2 * v]     = XH[w][colq][kb];
                afr[kk][2 * v + 1] = XH[w][colq][kb + 1];
            }
        }

        // ---- gates = [x h] @ [W_ih; W_hh] + b : 16 rows x 256 cols x K=128 ----
        #pragma unroll
        for (int j = 0; j < 4; ++j) {          // d-group: hidden cols j*16..j*16+15
            v8f acc[4];                        // i, f, g, o accumulator tiles
            #pragma unroll
            for (int g = 0; g < 4; ++g) {
                float bv = bsum[g * 64 + j * 16 + colq];
                #pragma unroll
                for (int v = 0; v < 8; ++v) acc[g][v] = bv;
            }
            #pragma unroll
            for (int kk = 0; kk < 4; ++kk) {   // K chunks of 32
                #pragma unroll
                for (int g = 0; g < 4; ++g) {
                    // B fragment: 32x16 bf16 (col = lane&15, K-half by lane>>4)
                    int col = g * 64 + j * 16 + colq;
                    int kb0 = kk * 32 + (khalf << 3);
                    v16bf bb;
                    #pragma unroll
                    for (int v = 0; v < 8; ++v) {
                        int kv = kb0 + ((v < 4) ? 2 * v : 16 + 2 * (v - 4));
                        bb[2 * v]     = WldsT[col][kv];
                        bb[2 * v + 1] = WldsT[col][kv + 1];
                    }
                    acc[g] = __builtin_amdgcn_wmma_f32_16x16x32_bf16(
                        false, afr[kk], false, bb, (short)0, acc[g], false, false);
                }
            }
            // ---- LSTM nonlinearity (torch gate order i,f,g,o) ----
            #pragma unroll
            for (int v = 0; v < 8; ++v) {
                float iv = acc[0][v], fv = acc[1][v], gv = acc[2][v], ov = acc[3][v];
                float c  = sigf(fv) * cst[j][v] + sigf(iv) * tanhfast(gv);
                cst[j][v]  = c;
                hnew[j][v] = sigf(ov) * tanhfast(c);
            }
        }
        // ---- publish h for next step (after ALL j consumed old h) ----
        #pragma unroll
        for (int j = 0; j < 4; ++j)
            #pragma unroll
            for (int v = 0; v < 8; ++v)
                XH[w][v + (khalf << 3)][64 + j * 16 + colq] = (__bf16)hnew[j][v];
    }

    // ---- write final hidden e = h_T (f32) ----
    #pragma unroll
    for (int j = 0; j < 4; ++j)
        #pragma unroll
        for (int v = 0; v < 8; ++v)
            e_out[(size_t)(rowbase + v + (khalf << 3)) * D_DIM + j * 16 + colq] = hnew[j][v];
}

// ---------------------------------------------------------------------------
// Kernel 2: heads + Gumbel-max sampling + rel/abs join. 1 thread = 1 batch row.
// ---------------------------------------------------------------------------
__device__ __forceinline__ float u01hash(uint64_t x) {
    x ^= 0x2545F4914F6CDD1Dull;
    x += 0x9E3779B97F4A7C15ull;
    x = (x ^ (x >> 30)) * 0xBF58476D1CE4E5B9ull;
    x = (x ^ (x >> 27)) * 0x94D049BB133111EBull;
    x ^= x >> 31;
    return ((float)(uint32_t)(x >> 40) + 0.5f) * (1.0f / 16777216.0f);
}

template <int C>
__device__ __forceinline__ void sample_head(const float* l, int b, int head,
                                            int& s, float& lp) {
    float m = l[0];
    #pragma unroll
    for (int c = 1; c < C; ++c) m = fmaxf(m, l[c]);
    float z = 0.0f;
    #pragma unroll
    for (int c = 0; c < C; ++c) z += __expf(l[c] - m);
    float lse = __logf(z);
    float best = -1e30f; int bi = 0;
    #pragma unroll
    for (int c = 0; c < C; ++c) {
        float u = u01hash(((uint64_t)head << 36) ^ ((uint64_t)b << 6) ^ (uint64_t)c ^ 42ull);
        float gm = -__logf(-__logf(u));
        float v = l[c] + gm;
        if (v > best) { best = v; bi = c; }
    }
    s  = bi;
    lp = l[bi] - m - lse;
}

__global__ __launch_bounds__(256) void heads_kernel(
    const float* __restrict__ e_ws,      // [B, 64]
    const float* __restrict__ canvas,    // [B, 25, 4]
    const int*   __restrict__ inst_type, // [B]
    const float* __restrict__ Wc, const float* __restrict__ bc,
    const float* __restrict__ Ws, const float* __restrict__ bs,
    const float* __restrict__ Wa, const float* __restrict__ ba,
    const float* __restrict__ Wl, const float* __restrict__ bl,
    const float* __restrict__ Wr, const float* __restrict__ br,
    const float* __restrict__ Wr1, const float* __restrict__ br1,
    const float* __restrict__ Wr2, const float* __restrict__ br2,
    float* __restrict__ out, int B)
{
    __shared__ float sWc[64 * 3], sWs[64 * 3], sWa[64 * 2], sWl[64 * 25], sWr[64 * 8];
    __shared__ float sWr1[68 * 32], sWr2[32];
    __shared__ float sbc[3], sbs[3], sba[2], sbl[25], sbr[8], sbr1[32], sbr2[1];

    const int tid = threadIdx.x;
    for (int i = tid; i < 64 * 3;  i += 256) sWc[i]  = Wc[i];
    for (int i = tid; i < 64 * 3;  i += 256) sWs[i]  = Ws[i];
    for (int i = tid; i < 64 * 2;  i += 256) sWa[i]  = Wa[i];
    for (int i = tid; i < 64 * 25; i += 256) sWl[i]  = Wl[i];
    for (int i = tid; i < 64 * 8;  i += 256) sWr[i]  = Wr[i];
    for (int i = tid; i < 68 * 32; i += 256) sWr1[i] = Wr1[i];
    for (int i = tid; i < 32;      i += 256) sWr2[i] = Wr2[i];
    if (tid < 3)  sbc[tid] = bc[tid];
    if (tid < 3)  sbs[tid] = bs[tid];
    if (tid < 2)  sba[tid] = ba[tid];
    if (tid < 25) sbl[tid] = bl[tid];
    if (tid < 8)  sbr[tid] = br[tid];
    if (tid < 32) sbr1[tid] = br1[tid];
    if (tid == 0) sbr2[0] = br2[0];
    __syncthreads();

    const int b = blockIdx.x * blockDim.x + tid;
    if (b >= B) return;

    float e[64];
    #pragma unroll
    for (int d = 0; d < 64; ++d) e[d] = e_ws[(size_t)b * 64 + d];

    // small linear heads
    float lc[3], lsv[3], la[2], ll[25], lr[8];
    #pragma unroll
    for (int c = 0; c < 3; ++c)  { float s = sbc[c]; for (int k = 0; k < 64; ++k) s += e[k] * sWc[k * 3 + c];  lc[c]  = s; }
    #pragma unroll
    for (int c = 0; c < 3; ++c)  { float s = sbs[c]; for (int k = 0; k < 64; ++k) s += e[k] * sWs[k * 3 + c];  lsv[c] = s; }
    #pragma unroll
    for (int c = 0; c < 2; ++c)  { float s = sba[c]; for (int k = 0; k < 64; ++k) s += e[k] * sWa[k * 2 + c];  la[c]  = s; }
    for (int c = 0; c < 25; ++c) { float s = sbl[c]; for (int k = 0; k < 64; ++k) s += e[k] * sWl[k * 25 + c]; ll[c]  = s; }
    #pragma unroll
    for (int c = 0; c < 8; ++c)  { float s = sbr[c]; for (int k = 0; k < 64; ++k) s += e[k] * sWr[k * 8 + c];  lr[c]  = s; }

    int color_s, shape_s, act_s, abs_s, rel_s, att_s;
    float color_lp, shape_lp, act_lp, abs_lp, rel_lp, att_lp;
    sample_head<3>(lc, b, 0, color_s, color_lp);
    sample_head<3>(lsv, b, 1, shape_s, shape_lp);
    sample_head<2>(la, b, 2, act_s, act_lp);
    sample_head<25>(ll, b, 3, abs_s, abs_lp);
    sample_head<8>(lr, b, 4, rel_s, rel_lp);

    // attention MLP: epart[h] shared across the 25 cells (e is constant per row)
    float ep[32];
    #pragma unroll
    for (int h = 0; h < 32; ++h) {
        float s = sbr1[h];
        for (int k = 0; k < 64; ++k) s += e[k] * sWr1[k * 32 + h];
        ep[h] = s;
    }
    float al[25];
    const float4* cvp = reinterpret_cast<const float4*>(canvas);
    for (int cell = 0; cell < 25; ++cell) {
        float4 cv = cvp[(size_t)b * 25 + cell];
        float s = sbr2[0];
        #pragma unroll
        for (int h = 0; h < 32; ++h) {
            float hv = ep[h] + cv.x * sWr1[64 * 32 + h] + cv.y * sWr1[65 * 32 + h]
                             + cv.z * sWr1[66 * 32 + h] + cv.w * sWr1[67 * 32 + h];
            hv = fmaxf(hv, 0.0f);
            s += hv * sWr2[h];
        }
        al[cell] = s;
    }
    sample_head<25>(al, b, 5, att_s, att_lp);

    float4 ro = cvp[(size_t)b * 25 + att_s];
    int rr = (int)ro.z, cc = (int)ro.w;
    const int offr[8] = {-1, 1, 0, 0, -1, -1, 1, 1};
    const int offc[8] = {0, 0, 1, -1, -1, 1, -1, 1};
    int rel_loc = (rr + offr[rel_s]) * 5 + (cc + offc[rel_s]);

    bool is_rel = (inst_type[b] - 1) == 1;
    int   loc    = is_rel ? rel_loc : abs_s;
    float loc_lp = is_rel ? rel_lp : abs_lp;
    float attlpo = is_rel ? att_lp : 0.0f;

    out[(size_t)b * 5 + 0] = color_lp;
    out[(size_t)b * 5 + 1] = shape_lp;
    out[(size_t)b * 5 + 2] = act_lp;
    out[(size_t)b * 5 + 3] = loc_lp;
    out[(size_t)b * 5 + 4] = attlpo;
    out[(size_t)B * 5 + b] = (float)loc;
}

// ---------------------------------------------------------------------------
extern "C" void kernel_launch(void* const* d_in, const int* in_sizes, int n_in,
                              void* d_out, int out_size, void* d_ws, size_t ws_size,
                              hipStream_t stream) {
    const int*   inst      = (const int*)  d_in[0];
    const float* canvas    = (const float*)d_in[1];
    const int*   inst_type = (const int*)  d_in[2];
    const float* embed     = (const float*)d_in[3];
    const float* W_ih      = (const float*)d_in[4];
    const float* W_hh      = (const float*)d_in[5];
    const float* b_ih      = (const float*)d_in[6];
    const float* b_hh      = (const float*)d_in[7];
    const float* Wc  = (const float*)d_in[8];  const float* bc  = (const float*)d_in[9];
    const float* Ws  = (const float*)d_in[10]; const float* bs  = (const float*)d_in[11];
    const float* Wa  = (const float*)d_in[12]; const float* ba  = (const float*)d_in[13];
    const float* Wl  = (const float*)d_in[14]; const float* bl  = (const float*)d_in[15];
    const float* Wr  = (const float*)d_in[16]; const float* br  = (const float*)d_in[17];
    const float* Wr1 = (const float*)d_in[18]; const float* br1 = (const float*)d_in[19];
    const float* Wr2 = (const float*)d_in[20]; const float* br2 = (const float*)d_in[21];

    const int B = in_sizes[0] / T_LEN;     // 16384
    float* e_ws = (float*)d_ws;            // [B, 64] f32 = 4 MB scratch

    lstm_wmma_kernel<<<B / 64, 128, 0, stream>>>(
        inst, embed, W_ih, W_hh, b_ih, b_hh, e_ws);

    heads_kernel<<<(B + 255) / 256, 256, 0, stream>>>(
        e_ws, canvas, inst_type, Wc, bc, Ws, bs, Wa, ba, Wl, bl,
        Wr, br, Wr1, br1, Wr2, br2, (float*)d_out, B);
}
